// NeuromorphicProcessor_61314953117856
// MI455X (gfx1250) — matrix operand order repaired and checked by reading, not compiled, not run on hardware
//
#include <hip/hip_runtime.h>

typedef __attribute__((ext_vector_type(16))) __bf16 v16bf;
typedef __attribute__((ext_vector_type(8)))  __bf16 v8bf;
typedef __attribute__((ext_vector_type(8)))  float  v8f;

#define T_STEPS 1000
#define NN      128   // neurons
#define DD      64    // input dim
#define BB      1000  // batch rows (== T in this reference)
#define KTILES  4     // 128 / 32 K tiles for bf16 WMMA

// ---------------------------------------------------------------------------
// Kernel 1: inp_cur[t][n] = sum_d input_signal[t][d] * input_weights[d][n]
// ---------------------------------------------------------------------------
__global__ void snn_input_gemm(const float* __restrict__ sig,
                               const float* __restrict__ w_in,
                               float* __restrict__ xcur) {
  const int t = blockIdx.x;     // 0..999
  const int n = threadIdx.x;    // 0..127
  float acc = 0.0f;
  #pragma unroll
  for (int d = 0; d < DD; ++d)
    acc = __builtin_fmaf(sig[t * DD + d], w_in[d * NN + n], acc);
  xcur[t * NN + n] = acc;
}

// ---------------------------------------------------------------------------
// Kernel 2: sequential scan. One 256-thread block (8 waves) per 16-row group.
// Wave w owns N-tile w: 4x v_wmma_f32_16x16x32_bf16 per step against the
// shared 16x128 bf16 trace (rebuilt from LDS in A layout each step).
// Trace LDS is ping-pong double-buffered -> single barrier per step.
// ---------------------------------------------------------------------------
__global__ __launch_bounds__(256, 1) void snn_scan(
    const float* __restrict__ w_rec,
    const float* __restrict__ xcur,
    float* __restrict__ spikes_out,   // (B, T, N)
    float* __restrict__ mem_out,      // (B, T, N)
    float* __restrict__ trace_out) {  // (B, N)
  __shared__ __align__(16) __bf16 ltr[2][16 * NN];  // ping-pong trace buffers

  const int tid   = threadIdx.x;
  const int lane  = tid & 31;
  const int nt    = tid >> 5;        // wave id == owned N-tile (0..7)
  const int half  = lane >> 4;       // 0/1
  const int nl    = lane & 15;       // col-in-tile (B/C/D layouts); row M (A)
  const int group = blockIdx.x;      // 16-row group

  const int colbase = nt * 16 + nl;  // this lane's global column n

  // ---- W fragments for this wave's N-tile (B layout: K = 16*half + j). ----
  v16bf Wf[KTILES];
  #pragma unroll
  for (int kt = 0; kt < KTILES; ++kt) {
    #pragma unroll
    for (int j = 0; j < 16; ++j)
      Wf[kt][j] = (__bf16)w_rec[(kt * 32 + half * 16 + j) * NN + colbase];
  }

  // ---- State in C/D layout: element r -> row m = r + 8*half, col = colbase.
  v8f syn = 0.0f, mem = 0.0f, refr = 0.0f, tr = 0.0f;

  for (int i = tid; i < 2 * 16 * NN; i += 256)
    ltr[0][i] = (__bf16)0.0f;        // zeros both buffers (flat)
  __syncthreads();

  // Output element offset for (b = group*16 + 8*half + 0, t = 0, n = colbase);
  // per-r delta r*T*N folds into the store immediate, per-step delta is +NN.
  unsigned obase = (unsigned)(group * 16 + 8 * half) * (T_STEPS * NN) + colbase;
  const int brow0 = group * 16 + 8 * half;

  for (int t = 0; t < T_STEPS; ++t) {
    const __bf16* rbuf = ltr[t & 1];
    __bf16*       wbuf = ltr[(t & 1) ^ 1];

    // ---- A fragments: slot j -> K=32kt+16h+j (matches W fragments);
    //      row M = nl. Two b128 loads each.
    v16bf A[KTILES];
    #pragma unroll
    for (int kt = 0; kt < KTILES; ++kt) {
      const __bf16* src = &rbuf[nl * NN + kt * 32 + half * 16];
      const v8bf lo = *(const v8bf*)(src);
      const v8bf hi = *(const v8bf*)(src + 8);
      #pragma unroll
      for (int j = 0; j < 8; ++j) { A[kt][j] = lo[j]; A[kt][j + 8] = hi[j]; }
    }

    // ---- rec tile for this wave's 16 columns, fp32 accumulate ----
    v8f acc = {};
    #pragma unroll
    for (int kt = 0; kt < KTILES; ++kt) {
      acc = __builtin_amdgcn_wmma_f32_16x16x32_bf16(
          false, A[kt], false, Wf[kt], (short)0, acc, false, false);
    }

    const float x = xcur[t * NN + colbase];  // broadcasts over rows

    #pragma unroll
    for (int r = 0; r < 8; ++r) {
      float s  = syn[r] * 0.8f + x + acc[r];      // syn*(1-1/5) + x + rec
      float mv = mem[r] * 0.9f + s * 0.1f;        // mem*(1-1/10) + syn/10
      float rf = refr[r];
      mv = (rf > 0.0f) ? 0.0f : mv;               // refractory mask
      rf = fmaxf(rf - 1.0f, 0.0f);
      const float sp = (mv > 1.0f) ? 1.0f : 0.0f; // spike
      mv = mv * (1.0f - sp);                      // reset on spike
      rf = rf + sp * 2.0f;                        // refractory period
      const float tv = tr[r] * 0.95f + sp;        // trace decay + spike
      syn[r] = s; mem[r] = mv; refr[r] = rf; tr[r] = tv;

      wbuf[(r + 8 * half) * NN + colbase] = (__bf16)tv;  // next step's A data

      if (brow0 + r < BB) {
        const unsigned o = obase + (unsigned)(r * T_STEPS * NN);
        __builtin_nontemporal_store(sp, spikes_out + o);
        __builtin_nontemporal_store(mv, mem_out + o);
      }
    }
    obase += NN;

    __syncthreads();  // trace stores visible before next step's A loads
  }

  // ---- final trace output (B, N) ----
  #pragma unroll
  for (int r = 0; r < 8; ++r) {
    if (brow0 + r < BB)
      __builtin_nontemporal_store(
          tr[r], trace_out + (unsigned)(brow0 + r) * NN + colbase);
  }
}

// ---------------------------------------------------------------------------
extern "C" void kernel_launch(void* const* d_in, const int* in_sizes, int n_in,
                              void* d_out, int out_size, void* d_ws, size_t ws_size,
                              hipStream_t stream) {
  const float* sig   = (const float*)d_in[0];  // (1000, 64)
  const float* w_in  = (const float*)d_in[1];  // (64, 128)
  const float* w_rec = (const float*)d_in[2];  // (128, 128)
  // d_in[3] = n_timesteps (scalar int) -- fixed at 1000, baked in.

  float* xcur = (float*)d_ws;                  // (1000, 128) input currents

  float* spikes = (float*)d_out;                       // (B,T,N)
  float* memh   = spikes + (size_t)BB * T_STEPS * NN;  // (B,T,N)
  float* traceo = memh   + (size_t)BB * T_STEPS * NN;  // (B,N)

  snn_input_gemm<<<T_STEPS, NN, 0, stream>>>(sig, w_in, xcur);

  const int ngroups = (BB + 15) / 16;  // 63 groups, 8 waves each
  snn_scan<<<ngroups, 256, 0, stream>>>(w_rec, xcur, spikes, memh, traceo);
}
